// Classifier2_54022098649409
// MI455X (gfx1250) — matrix operand order repaired and checked by reading, not compiled, not run on hardware
//
#include <hip/hip_runtime.h>
#include <hip/hip_fp16.h>

// ---------------------------------------------------------------------------
// GNN classifier pipeline for MI455X (gfx1250), wave32 + WMMA f16->f32.
//   TDM(x tile -> LDS) -> conv(dil 1/2/3, fused epilogue, WMMA from LDS)
//   -> 3x gconv(96->54) -> gconv(54->25) -> gconv(25->13), gconv(25->11)
//   -> segment sums -> tiny MLP.
// Weights are pre-swizzled to [out][in] f16 with zero padding so every B
// fragment load is a contiguous b128 and no guards sit inside WMMA loops.
// ---------------------------------------------------------------------------

typedef __attribute__((ext_vector_type(16))) _Float16 v16h;
typedef __attribute__((ext_vector_type(8)))  _Float16 v8h;
typedef __attribute__((ext_vector_type(8)))  float    v8f;
typedef __attribute__((ext_vector_type(4)))  unsigned int u32x4;
typedef __attribute__((ext_vector_type(8)))  int       i32x8;
typedef __attribute__((ext_vector_type(4)))  int       i32x4;

#define NNODES 51200
#define NEDGES 819200
#define NGRAPH 256

// conv LDS tile: 64 output rows + 6 halo rows, 128 features, f16.
#define XT_ROWS 70

// ---------------- WMMA helpers --------------------------------------------

__device__ __forceinline__ v8f wmma16(v16h a, v16h b, v8f c) {
  // D = A(16x32 f16) * B(32x16 f16) + C(16x16 f32)
  return __builtin_amdgcn_wmma_f32_16x16x32_f16(
      /*neg_a=*/false, a, /*neg_b=*/false, b,
      /*c_mod=*/(short)0, c, /*reuse_a=*/false, /*reuse_b=*/false);
}

// A fragment from row-major f16 [rows][ldk] in GLOBAL memory.
__device__ __forceinline__ v16h load_a16(const _Float16* base, int row0,
                                         int ldk, int kbase) {
  const int lane = threadIdx.x & 31;
  const int m = lane & 15, hi = lane >> 4;
  const _Float16* p = base + (size_t)(row0 + m) * ldk + kbase + hi * 8;
  union { v16h v; v8h h[2]; } u;
  u.h[0] = *(const v8h*)p;
  u.h[1] = *(const v8h*)(p + 16);
  return u.v;
}

// B fragment from pre-transposed weights Bt[out][in] (f16): 32x16 tile,
// column n = col0 + (lane&15), K = kbase + (lane>>4)*8 + {0..7, 16..23}.
__device__ __forceinline__ v16h load_b16(const _Float16* bt, int col0,
                                         int ldk, int kbase) {
  const int lane = threadIdx.x & 31;
  const int n = lane & 15, hi = lane >> 4;
  const _Float16* p = bt + (size_t)(col0 + n) * ldk + kbase + hi * 8;
  union { v16h v; v8h h[2]; } u;
  u.h[0] = *(const v8h*)p;
  u.h[1] = *(const v8h*)(p + 16);
  return u.v;
}

// D/C f32 16x16 store: lane holds col = lane&15, rows (lane>>4)*8 + r.
__device__ __forceinline__ void store_d_f32(float* out, int row0, int ldn,
                                            int col0, v8f acc) {
  const int lane = threadIdx.x & 31;
  const int col = col0 + (lane & 15);
  const int rb = row0 + ((lane >> 4) * 8);
#pragma unroll
  for (int r = 0; r < 8; ++r) out[(size_t)(rb + r) * ldn + col] = acc[r];
}

// ---------------- prep kernels --------------------------------------------

__global__ void pad_x_kernel(const float* x, _Float16* xpad) {
  int idx = blockIdx.x * blockDim.x + threadIdx.x;
  const int total = (NNODES + 6) * 128;
  if (idx >= total) return;
  int r = idx >> 7, c = idx & 127;
  float v = (r >= 3 && r < NNODES + 3) ? x[(size_t)(r - 3) * 128 + c] : 0.f;
  xpad[idx] = (_Float16)v;
}

// conv weights [96,128,3] f32 -> Bconv[(d*3+tap)][o][i] f16 (ldk = 128).
__global__ void pack_convw_kernel(const float* w1, const float* w2,
                                  const float* w3, _Float16* bconv) {
  int idx = blockIdx.x * blockDim.x + threadIdx.x;
  const int total = 9 * 96 * 128;
  if (idx >= total) return;
  int i = idx & 127;
  int o = (idx >> 7) % 96;
  int dt = idx / (96 * 128);           // 0..8
  int d = dt / 3, tap = dt % 3;
  const float* w = (d == 0) ? w1 : (d == 1) ? w2 : w3;
  bconv[idx] = (_Float16)w[(size_t)o * 384 + i * 3 + tap];
}

__global__ void pack_w1_kernel(const float* W11, const float* W12,
                               const float* W13, _Float16* w1t) {
  int idx = blockIdx.x * blockDim.x + threadIdx.x;
  const int total = 3 * 64 * 96;
  if (idx >= total) return;
  int i = idx % 96;
  int o = (idx / 96) % 64;
  int j = idx / (64 * 96);
  const float* W = (j == 0) ? W11 : (j == 1) ? W12 : W13;
  w1t[idx] = (_Float16)((o < 54) ? W[(size_t)i * 54 + o] : 0.f);
}

__global__ void pack_w2_kernel(const float* W2, _Float16* w2t) {
  int idx = blockIdx.x * blockDim.x + threadIdx.x;
  if (idx >= 32 * 64) return;
  int i = idx & 63, o = idx >> 6;
  w2t[idx] = (_Float16)((o < 25 && i < 54) ? W2[(size_t)i * 25 + o] : 0.f);
}
__global__ void pack_w34_kernel(const float* W3, const float* W4,
                                _Float16* w3t, _Float16* w4t) {
  int idx = blockIdx.x * blockDim.x + threadIdx.x;
  if (idx >= 16 * 32) return;
  int i = idx & 31, o = idx >> 5;
  w3t[idx] = (_Float16)((o < 13 && i < 25) ? W3[(size_t)i * 13 + o] : 0.f);
  w4t[idx] = (_Float16)((o < 11 && i < 25) ? W4[(size_t)i * 11 + o] : 0.f);
}

// ---------------- fused dilated-conv (TDM -> LDS -> WMMA) ------------------
// Workgroup of 4 waves covers 64 rows. The TDM loads a 70x128 f16 halo tile
// of xpad into LDS once; all A fragments then come from LDS (ds_load_b128).
__global__ void __launch_bounds__(128)
conv_wmma_kernel(const _Float16* __restrict__ xpad,
                 const _Float16* __restrict__ bconv,
                 const float* __restrict__ cb1, const float* __restrict__ cb2,
                 const float* __restrict__ cb3, _Float16* __restrict__ h0buf) {
  __shared__ _Float16 xtile[XT_ROWS * 128];  // only LDS object -> offset 0
  const int wave = threadIdx.x >> 5;
  const int lane = threadIdx.x & 31;
  const int blockRow = blockIdx.x * 64;      // xpad row index of tile start

#if __has_builtin(__builtin_amdgcn_tensor_load_to_lds)
  if (wave == 0) {
    // Tensor DMA descriptor (D#), 2D tile of the [51206,128] f16 tensor.
    const unsigned long long ga =
        (unsigned long long)(const void*)(xpad + (size_t)blockRow * 128);
    u32x4 g0;
    g0[0] = 1u;                                        // count=1, user mode
    g0[1] = 0u;                                        // lds_addr = xtile @ 0
    g0[2] = (unsigned)(ga & 0xFFFFFFFFu);              // global_addr[31:0]
    g0[3] = (unsigned)((ga >> 32) & 0x01FFFFFFu)       // global_addr[56:32]
            | 0x80000000u;                             // type = 2 ("image")
    i32x8 g1;
    g1[0] = (int)(1u << 16);                           // data_size = 2 bytes
    g1[1] = (int)(128u << 16);                         // tensor_dim0 = 128
    g1[2] = (int)(((unsigned)(NNODES + 6) & 0xFFFFu) << 16);  // tensor_dim1 lo
    g1[3] = (int)(128u << 16);                         // dim1 hi=0 | tile_dim0
    g1[4] = XT_ROWS;                                   // tile_dim1 = 70
    g1[5] = 128;                                       // tensor_dim0_stride
    g1[6] = 0;
    g1[7] = 0;
    i32x4 z4 = {0, 0, 0, 0};                           // 2D: groups 2/3 unused
    i32x8 z8 = {0, 0, 0, 0, 0, 0, 0, 0};
    // 6-arg form (this toolchain): (g0, g1, g2, g3, gx, cpol)
    __builtin_amdgcn_tensor_load_to_lds(g0, g1, z4, z4, z8, 0);
    __builtin_amdgcn_s_wait_tensorcnt(0);
  }
#else
  // Fallback: cooperative b128 copy global -> LDS.
  {
    const v8h* gsrc = (const v8h*)(xpad + (size_t)blockRow * 128);
    v8h* ldst = (v8h*)xtile;
    for (int i = threadIdx.x; i < (XT_ROWS * 128) / 8; i += 128)
      ldst[i] = gsrc[i];
  }
#endif
  // The TDM builtin carries no pointer operand, so the compiler cannot see
  // that it wrote xtile. Escape the LDS tile into an opaque asm with a
  // memory clobber so the allocation and the ds_load_b128 reads stay live.
  {
    void* xesc = (void*)xtile;
    asm volatile("" : "+v"(xesc) :: "memory");
  }
  __syncthreads();

  const int row0 = blockRow + wave * 16;     // output row base
  const int colL = lane & 15;
  const int hi8 = (lane >> 4) * 8;
  const int m = lane & 15, hi = lane >> 4;
  const size_t NH = (size_t)NNODES * 96;

  // A fragment straight from the LDS tile (two ds_load_b128 per fragment).
  auto load_a_lds = [&](int lrow0, int kbase) -> v16h {
    const _Float16* p = &xtile[(lrow0 + m) * 128 + kbase + hi * 8];
    union { v16h v; v8h h[2]; } u;
    u.h[0] = *(const v8h*)p;
    u.h[1] = *(const v8h*)(p + 16);
    return u.v;
  };

  for (int ct = 0; ct < 6; ++ct) {
    v8f acc[3];
#pragma unroll
    for (int d = 0; d < 3; ++d) acc[d] = v8f{0, 0, 0, 0, 0, 0, 0, 0};

#pragma unroll
    for (int d = 0; d < 3; ++d) {
#pragma unroll
      for (int tap = 0; tap < 3; ++tap) {
        const int lrow0 = wave * 16 + 3 + (tap - 1) * (d + 1);  // 0..54
        const _Float16* B = bconv + (size_t)(d * 3 + tap) * 96 * 128;
#pragma unroll
        for (int ks = 0; ks < 4; ++ks) {
          v16h a = load_a_lds(lrow0, ks * 32);
          v16h b = load_b16(B, ct * 16, 128, ks * 32);
          acc[d] = wmma16(a, b, acc[d]);
        }
      }
    }
    const float bb1 = cb1[ct * 16 + colL];
    const float bb2 = cb2[ct * 16 + colL];
    const float bb3 = cb3[ct * 16 + colL];
#pragma unroll
    for (int r = 0; r < 8; ++r) {
      const float c1 = acc[0][r] + bb1;
      const float c2 = acc[1][r] + bb2;
      const float c3 = acc[2][r] + bb3;
      const size_t off = (size_t)(row0 + hi8 + r) * 96 + ct * 16 + colL;
      h0buf[0 * NH + off] = (_Float16)(fmaxf(c1, 0.f) + c2);
      h0buf[1 * NH + off] = (_Float16)(fmaxf(c2, 0.f) + c3);
      h0buf[2 * NH + off] = (_Float16)(fmaxf(c3, 0.f) + c1);
    }
  }
}

// ---------------- gconv GEMMs (WMMA) --------------------------------------

// hw_j = h0_j[N,96] @ W1t_j -> [N,64] f32 (cols >=54 are zero by padding).
__global__ void __launch_bounds__(128)
gemm1_kernel(const _Float16* __restrict__ h0buf,
             const _Float16* __restrict__ w1t, float* __restrict__ hw1) {
  const int tile = blockIdx.x * 4 + (threadIdx.x >> 5);
  const int row0 = tile * 16;
  const size_t NH = (size_t)NNODES * 96;
  const size_t NO = (size_t)NNODES * 64;
#pragma unroll
  for (int j = 0; j < 3; ++j) {
    const _Float16* A = h0buf + j * NH;
    const _Float16* B = w1t + (size_t)j * 64 * 96;
    float* O = hw1 + j * NO;
    for (int ct = 0; ct < 4; ++ct) {
      v8f acc = v8f{0, 0, 0, 0, 0, 0, 0, 0};
#pragma unroll
      for (int ks = 0; ks < 3; ++ks)
        acc = wmma16(load_a16(A, row0, 96, ks * 32),
                     load_b16(B, ct * 16, 96, ks * 32), acc);
      store_d_f32(O, row0, 64, ct * 16, acc);
    }
  }
}

// hw2 = ach1[N,64] @ W2t -> [N,32] f32
__global__ void __launch_bounds__(128)
gemm2_kernel(const _Float16* __restrict__ ach1h,
             const _Float16* __restrict__ w2t, float* __restrict__ hw2) {
  const int tile = blockIdx.x * 4 + (threadIdx.x >> 5);
  const int row0 = tile * 16;
#pragma unroll
  for (int ct = 0; ct < 2; ++ct) {
    v8f acc = v8f{0, 0, 0, 0, 0, 0, 0, 0};
#pragma unroll
    for (int ks = 0; ks < 2; ++ks)
      acc = wmma16(load_a16(ach1h, row0, 64, ks * 32),
                   load_b16(w2t, ct * 16, 64, ks * 32), acc);
    store_d_f32(hw2, row0, 32, ct * 16, acc);
  }
}

// hw3 = ach2[N,32] @ W3t -> [N,16];  hw4 = ach2 @ W4t -> [N,16]
__global__ void __launch_bounds__(128)
gemm34_kernel(const _Float16* __restrict__ ach2h,
              const _Float16* __restrict__ w3t,
              const _Float16* __restrict__ w4t,
              float* __restrict__ hw3, float* __restrict__ hw4) {
  const int tile = blockIdx.x * 4 + (threadIdx.x >> 5);
  const int row0 = tile * 16;
  v16h a = load_a16(ach2h, row0, 32, 0);
  v8f z = v8f{0, 0, 0, 0, 0, 0, 0, 0};
  v8f acc3 = wmma16(a, load_b16(w3t, 0, 32, 0), z);
  v8f acc4 = wmma16(a, load_b16(w4t, 0, 32, 0), z);
  store_d_f32(hw3, row0, 16, 0, acc3);
  store_d_f32(hw4, row0, 16, 0, acc4);
}

// ---------------- edge scatter (weighted segment-sum) ----------------------

__global__ void scatter1_kernel(const int* __restrict__ src,
                                const int* __restrict__ dst,
                                const float* __restrict__ ew,
                                const float* __restrict__ hw1,
                                float* __restrict__ agg1) {
  long long tid = (long long)blockIdx.x * blockDim.x + threadIdx.x;
  if (tid >= (long long)NEDGES * 64) return;
  int e = (int)(tid >> 6), f = (int)(tid & 63);
  int s = src[e], t = dst[e];
  float w = ew[e];
  const size_t NO = (size_t)NNODES * 64;
#pragma unroll
  for (int j = 0; j < 3; ++j)
    atomicAdd(&agg1[j * NO + (size_t)t * 64 + f],
              hw1[j * NO + (size_t)s * 64 + f] * w);
}

__global__ void scatter2_kernel(const int* __restrict__ src,
                                const int* __restrict__ dst,
                                const float* __restrict__ ew,
                                const float* __restrict__ hw2,
                                float* __restrict__ agg2) {
  long long tid = (long long)blockIdx.x * blockDim.x + threadIdx.x;
  if (tid >= (long long)NEDGES * 32) return;
  int e = (int)(tid >> 5), f = (int)(tid & 31);
  atomicAdd(&agg2[(size_t)dst[e] * 32 + f],
            hw2[(size_t)src[e] * 32 + f] * ew[e]);
}

__global__ void scatter34_kernel(const int* __restrict__ src,
                                 const int* __restrict__ dst,
                                 const float* __restrict__ ew,
                                 const float* __restrict__ hw3,
                                 const float* __restrict__ hw4,
                                 float* __restrict__ agg3,
                                 float* __restrict__ agg4) {
  long long tid = (long long)blockIdx.x * blockDim.x + threadIdx.x;
  if (tid >= (long long)NEDGES * 16) return;
  int e = (int)(tid >> 4), f = (int)(tid & 15);
  int s = src[e], t = dst[e];
  float w = ew[e];
  atomicAdd(&agg3[(size_t)t * 16 + f], hw3[(size_t)s * 16 + f] * w);
  atomicAdd(&agg4[(size_t)t * 16 + f], hw4[(size_t)s * 16 + f] * w);
}

// ---------------- activations / combines -----------------------------------

__global__ void combine1_kernel(const float* __restrict__ agg1,
                                const float* __restrict__ b11,
                                const float* __restrict__ b12,
                                const float* __restrict__ b13,
                                _Float16* __restrict__ ach1h) {
  int tid = blockIdx.x * blockDim.x + threadIdx.x;
  if (tid >= NNODES * 64) return;
  int f = tid & 63;
  float v = 0.f;
  if (f < 54) {
    const size_t NO = (size_t)NNODES * 64;
    v = fmaxf(agg1[0 * NO + tid] + b11[f], 0.f) +
        fmaxf(agg1[1 * NO + tid] + b12[f], 0.f) +
        fmaxf(agg1[2 * NO + tid] + b13[f], 0.f);
  }
  ach1h[tid] = (_Float16)v;
}

__global__ void acth2_kernel(const float* __restrict__ agg2,
                             const float* __restrict__ b2,
                             float* __restrict__ ach2f,
                             _Float16* __restrict__ ach2h) {
  int tid = blockIdx.x * blockDim.x + threadIdx.x;
  if (tid >= NNODES * 32) return;
  int f = tid & 31;
  float v = (f < 25) ? fmaxf(agg2[tid] + b2[f], 0.f) : 0.f;
  ach2f[tid] = v;
  ach2h[tid] = (_Float16)v;
}

// ---------------- per-graph segment sum ------------------------------------

__global__ void segsum_kernel(const int* __restrict__ gid,
                              const float* __restrict__ ach2f,
                              const float* __restrict__ agg3,
                              const float* __restrict__ agg4,
                              const float* __restrict__ b3,
                              const float* __restrict__ b4,
                              float* __restrict__ hg) {
  int tid = blockIdx.x * blockDim.x + threadIdx.x;
  if (tid >= NNODES * 64) return;
  int n = tid >> 6, f = tid & 63;
  if (f >= 49) return;
  float v;
  if (f < 25)       v = ach2f[(size_t)n * 32 + f];
  else if (f < 38)  v = fmaxf(agg3[(size_t)n * 16 + (f - 25)] + b3[f - 25], 0.f);
  else              v = fmaxf(agg4[(size_t)n * 16 + (f - 38)] + b4[f - 38], 0.f);
  atomicAdd(&hg[(size_t)gid[n] * 49 + f], v);
}

// ---------------- readout MLP ----------------------------------------------

__global__ void mlp_kernel(const float* __restrict__ hg,
                           const float* __restrict__ Wc1,
                           const float* __restrict__ bc1,
                           const float* __restrict__ Wc2,
                           const float* __restrict__ bc2,
                           float* __restrict__ out) {
  int g = threadIdx.x;  // 256 threads, one graph each
  float h[49];
#pragma unroll
  for (int i = 0; i < 49; ++i) h[i] = hg[(size_t)g * 49 + i];
  float o5[5];
#pragma unroll
  for (int k = 0; k < 5; ++k) o5[k] = bc2[k];
  for (int o = 0; o < 27; ++o) {
    float t = bc1[o];
#pragma unroll
    for (int i = 0; i < 49; ++i) t += h[i] * Wc1[i * 27 + o];
#pragma unroll
    for (int k = 0; k < 5; ++k) o5[k] += t * Wc2[o * 5 + k];
  }
#pragma unroll
  for (int k = 0; k < 5; ++k) out[(size_t)g * 5 + k] = o5[k];
}

// ---------------- host launcher --------------------------------------------

static inline size_t alignUp(size_t v) { return (v + 255) & ~(size_t)255; }

extern "C" void kernel_launch(void* const* d_in, const int* in_sizes, int n_in,
                              void* d_out, int out_size, void* d_ws,
                              size_t ws_size, hipStream_t stream) {
  const float* x    = (const float*)d_in[0];
  const int* src    = (const int*)d_in[1];
  const int* dst    = (const int*)d_in[2];
  const int* gid    = (const int*)d_in[3];
  const float* ew   = (const float*)d_in[4];
  const float* c1w  = (const float*)d_in[5];
  const float* c1b  = (const float*)d_in[6];
  const float* c2w  = (const float*)d_in[7];
  const float* c2b  = (const float*)d_in[8];
  const float* c3w  = (const float*)d_in[9];
  const float* c3b  = (const float*)d_in[10];
  const float* W11  = (const float*)d_in[11];
  const float* b11  = (const float*)d_in[12];
  const float* W12  = (const float*)d_in[13];
  const float* b12  = (const float*)d_in[14];
  const float* W13  = (const float*)d_in[15];
  const float* b13  = (const float*)d_in[16];
  const float* W2   = (const float*)d_in[17];
  const float* b2   = (const float*)d_in[18];
  const float* W3   = (const float*)d_in[19];
  const float* b3   = (const float*)d_in[20];
  const float* W4   = (const float*)d_in[21];
  const float* b4   = (const float*)d_in[22];
  const float* Wc1  = (const float*)d_in[23];
  const float* bc1  = (const float*)d_in[24];
  const float* Wc2  = (const float*)d_in[25];
  const float* bc2  = (const float*)d_in[26];
  float* out = (float*)d_out;

  // workspace layout
  char* ws = (char*)d_ws;
  size_t off = 0;
  auto take = [&](size_t bytes) { char* p = ws + off; off += alignUp(bytes); return p; };
  _Float16* xpad  = (_Float16*)take((size_t)(NNODES + 6) * 128 * 2);
  _Float16* bconv = (_Float16*)take((size_t)9 * 96 * 128 * 2);
  _Float16* w1t   = (_Float16*)take((size_t)3 * 64 * 96 * 2);
  _Float16* w2t   = (_Float16*)take((size_t)32 * 64 * 2);
  _Float16* w3t   = (_Float16*)take((size_t)16 * 32 * 2);
  _Float16* w4t   = (_Float16*)take((size_t)16 * 32 * 2);
  _Float16* h0buf = (_Float16*)take((size_t)3 * NNODES * 96 * 2);
  float*    hw1   = (float*)take((size_t)3 * NNODES * 64 * 4);
  float*    agg1  = (float*)take((size_t)3 * NNODES * 64 * 4);
  _Float16* ach1h = (_Float16*)take((size_t)NNODES * 64 * 2);
  float*    hw2   = (float*)take((size_t)NNODES * 32 * 4);
  float*    agg2  = (float*)take((size_t)NNODES * 32 * 4);
  float*    ach2f = (float*)take((size_t)NNODES * 32 * 4);
  _Float16* ach2h = (_Float16*)take((size_t)NNODES * 32 * 2);
  float*    hw3   = (float*)take((size_t)NNODES * 16 * 4);
  float*    hw4   = (float*)take((size_t)NNODES * 16 * 4);
  float*    agg3  = (float*)take((size_t)NNODES * 16 * 4);
  float*    agg4  = (float*)take((size_t)NNODES * 16 * 4);
  float*    hg    = (float*)take((size_t)NGRAPH * 49 * 4);
  (void)ws_size; (void)n_in; (void)in_sizes; (void)out_size;

  // zero accumulators every call (graph-capture legal)
  (void)hipMemsetAsync(agg1, 0, (size_t)3 * NNODES * 64 * 4, stream);
  (void)hipMemsetAsync(agg2, 0, (size_t)NNODES * 32 * 4, stream);
  (void)hipMemsetAsync(agg3, 0, (size_t)NNODES * 16 * 4, stream);
  (void)hipMemsetAsync(agg4, 0, (size_t)NNODES * 16 * 4, stream);
  (void)hipMemsetAsync(hg,   0, (size_t)NGRAPH * 49 * 4, stream);

  const int T = 256;
  // prep
  pad_x_kernel<<<((NNODES + 6) * 128 + T - 1) / T, T, 0, stream>>>(x, xpad);
  pack_convw_kernel<<<(9 * 96 * 128 + T - 1) / T, T, 0, stream>>>(c1w, c2w, c3w, bconv);
  pack_w1_kernel<<<(3 * 64 * 96 + T - 1) / T, T, 0, stream>>>(W11, W12, W13, w1t);
  pack_w2_kernel<<<(32 * 64 + T - 1) / T, T, 0, stream>>>(W2, w2t);
  pack_w34_kernel<<<(16 * 32 + T - 1) / T, T, 0, stream>>>(W3, W4, w3t, w4t);

  const int NT = NNODES / 16;          // 3200 row tiles
  const int GB = NT / 4;               // 4 waves (64 rows) per block
  // fused dilated conv: TDM -> LDS -> WMMA
  conv_wmma_kernel<<<GB, 128, 0, stream>>>(xpad, bconv, c1b, c2b, c3b, h0buf);
  // gconv layer 1
  gemm1_kernel<<<GB, 128, 0, stream>>>(h0buf, w1t, hw1);
  scatter1_kernel<<<(int)(((long long)NEDGES * 64 + T - 1) / T), T, 0, stream>>>(
      src, dst, ew, hw1, agg1);
  combine1_kernel<<<(NNODES * 64 + T - 1) / T, T, 0, stream>>>(agg1, b11, b12, b13, ach1h);
  // gconv layer 2
  gemm2_kernel<<<GB, 128, 0, stream>>>(ach1h, w2t, hw2);
  scatter2_kernel<<<(int)(((long long)NEDGES * 32 + T - 1) / T), T, 0, stream>>>(
      src, dst, ew, hw2, agg2);
  acth2_kernel<<<(NNODES * 32 + T - 1) / T, T, 0, stream>>>(agg2, b2, ach2f, ach2h);
  // gconv layers 3 & 4
  gemm34_kernel<<<GB, 128, 0, stream>>>(ach2h, w3t, w4t, hw3, hw4);
  scatter34_kernel<<<(int)(((long long)NEDGES * 16 + T - 1) / T), T, 0, stream>>>(
      src, dst, ew, hw3, hw4, agg3, agg4);
  // readout
  segsum_kernel<<<(NNODES * 64 + T - 1) / T, T, 0, stream>>>(gid, ach2f, agg3, agg4, b3, b4, hg);
  mlp_kernel<<<1, NGRAPH, 0, stream>>>(hg, Wc1, bc1, Wc2, bc2, out);
}
